// EGHM_22771916603488
// MI455X (gfx1250) — matrix operand order repaired and checked by reading, not compiled, not run on hardware
//
#include <hip/hip_runtime.h>
#include <hip/hip_bf16.h>

// Shapes from the reference
#define CH   128     // C_H
#define CIN  64      // C_IN
#define HW   16384   // 128*128
#define NPIX 32768   // B*HW, B=2
#define WDIM 128
#define LDSLD 136    // padded LDS row stride in bf16 elems (128 + 8 -> 272B)

typedef __bf16 bf16_t;
typedef __bf16 v16bf __attribute__((ext_vector_type(16)));
typedef float  v8f   __attribute__((ext_vector_type(8)));
typedef unsigned u32x4 __attribute__((ext_vector_type(4)));
typedef int      i32x4 __attribute__((ext_vector_type(4)));
typedef int      i32x8 __attribute__((ext_vector_type(8)));

#if __has_builtin(__builtin_amdgcn_tensor_load_to_lds) && \
    __has_builtin(__builtin_amdgcn_s_wait_tensorcnt)
#define HAVE_TDM 1
#else
#define HAVE_TDM 0
#endif

__device__ __forceinline__ v8f zero8() {
  v8f z;
#pragma unroll
  for (int i = 0; i < 8; ++i) z[i] = 0.f;
  return z;
}

__device__ __forceinline__ v8f wmma_bf16(v16bf a, v16bf b, v8f c) {
  // D = A(16x32 bf16) * B(32x16 bf16) + C(16x16 f32)
  return __builtin_amdgcn_wmma_f32_16x16x32_bf16(false, a, false, b, (short)0, c,
                                                 false, false);
}

// A fragment from row-major W[m][k] (ld elements per row).
// wave32 layout: lane L holds row M=L&15; K = (L>>4)*16 + j, j = 0..15
__device__ __forceinline__ v16bf load_a(const bf16_t* Wm, int ld, int mtile,
                                        int k0, int lane) {
  const bf16_t* row = Wm + (size_t)(mtile * 16 + (lane & 15)) * ld + k0 +
                      ((lane >> 4) << 4);
  v16bf a;
#pragma unroll
  for (int j = 0; j < 8; ++j) a[j] = row[j];
#pragma unroll
  for (int j = 0; j < 8; ++j) a[8 + j] = row[8 + j];
  return a;
}

// B fragment from pixel-major activations act[n][c]: per-lane fully contiguous.
__device__ __forceinline__ v16bf load_b_px(const bf16_t* act, int n, int kt,
                                           int lane) {
  const bf16_t* p = act + (size_t)n * CH + kt * 32 + ((lane >> 4) << 4);
  v16bf b;
#pragma unroll
  for (int j = 0; j < 16; ++j) b[j] = p[j];
  return b;
}

#if HAVE_TDM
// Issue a TDM load of one 128x128 bf16 weight tile (32KB) into LDS with
// 16B padding after every 256B row (LDS row stride 272B = LDSLD elems).
// D# packing per CDNA5 ISA 8.3/8.4; 1-D tile of 16384 2-byte elements.
__device__ __forceinline__ void tdm_load_wk(const bf16_t* gsrc, void* lds_dst) {
  unsigned lds_off = (unsigned)(uintptr_t)lds_dst;  // LDS aperture: low 32 bits
  unsigned long long ga = (unsigned long long)(uintptr_t)gsrc;
  u32x4 g0;
  g0[0] = 1u;                                   // count=1, user descriptor
  g0[1] = lds_off;                              // lds_addr
  g0[2] = (unsigned)(ga & 0xffffffffu);         // global_addr[31:0]
  g0[3] = (unsigned)((ga >> 32) & 0x1ffffffu)   // global_addr[56:32]
          | (2u << 30);                         // type = 2 ("image")
  i32x8 g1;
  g1[0] = (1 << 16)                             // data_size=1 (2 bytes)
          | (1 << 20)                           // pad_enable
          | (5 << 22)                           // pad_interval: 64 DWORDs
          | (3 << 25);                          // pad_amount: 4 DWORDs (16B)
  g1[1] = (int)(16384u << 16);                  // tensor_dim0 = 16384 (lo16)
  g1[2] = (int)(1u << 16);                      // tensor_dim0 hi=0; tensor_dim1=1
  g1[3] = (int)(16384u << 16);                  // tensor_dim1 hi=0; tile_dim0=16384
  g1[4] = 0;                                    // tile_dim1=0, tile_dim2=0
  g1[5] = 16384;                                // tensor_dim0_stride (lo32)
  g1[6] = 0;
  g1[7] = 0;
  i32x4 z4 = {0, 0, 0, 0};
  i32x8 z8 = {0, 0, 0, 0, 0, 0, 0, 0};
  // 6-arg (clang-23 / therock) form: (g0, g1, g2, g3, g4, cpol)
  __builtin_amdgcn_tensor_load_to_lds(g0, g1, z4, z4, z8, 0);
}
#endif

// ---------------------------------------------------------------------------
// Weight repack: f32 -> bf16, with layout transposes for df1/df2 so the tap
// index becomes the outermost (GEMM-per-tap friendly) dimension.
// ---------------------------------------------------------------------------
__global__ void k_prep_weights(const float* __restrict__ w_in,
                               const float* __restrict__ w_df1,
                               const float* __restrict__ w_df2,
                               const float* __restrict__ w_sp,
                               const float* __restrict__ w_out,
                               bf16_t* __restrict__ wi, bf16_t* __restrict__ w1,
                               bf16_t* __restrict__ w2, bf16_t* __restrict__ ws,
                               bf16_t* __restrict__ wo) {
  int idx = blockIdx.x * blockDim.x + threadIdx.x;
  const int N_IN = CH * CIN;        // 8192
  const int N_DF1 = 9 * CH * CH;    // 147456
  const int N_DF2 = 25 * CH * CH;   // 409600
  const int N_SP = CH * CH;         // 16384
  const int N_OUT = CIN * CH;       // 8192
  if (idx < N_IN) { wi[idx] = (bf16_t)w_in[idx]; return; }
  idx -= N_IN;
  if (idx < N_DF1) {  // dest [tap][o][i] <- src [o][i][tap]
    int tap = idx / (CH * CH);
    int rem = idx % (CH * CH);
    int o = rem >> 7, i = rem & 127;
    w1[idx] = (bf16_t)w_df1[(size_t)(o * CH + i) * 9 + tap];
    return;
  }
  idx -= N_DF1;
  if (idx < N_DF2) {  // dest [k][c][i] <- src [(c*25+k)][i]
    int k = idx / (CH * CH);
    int rem = idx % (CH * CH);
    int c = rem >> 7, i = rem & 127;
    w2[idx] = (bf16_t)w_df2[(size_t)(c * 25 + k) * CH + i];
    return;
  }
  idx -= N_DF2;
  if (idx < N_SP) { ws[idx] = (bf16_t)w_sp[idx]; return; }
  idx -= N_SP;
  if (idx < N_OUT) { wo[idx] = (bf16_t)w_out[idx]; return; }
}

// ---------------------------------------------------------------------------
// Kernel A: feat = LayerNorm_c(w_in @ x)   for both modalities (blockIdx.y)
// One wave computes a [128 x 16]-pixel tile; LN fused via one lane shuffle.
// Intermediates stored pixel-major [n][c] so later fragment loads are b128s.
// ---------------------------------------------------------------------------
__global__ __launch_bounds__(128) void k_in_ln(
    const float* __restrict__ rgb, const float* __restrict__ evt,
    const bf16_t* __restrict__ w_in_bf, const float* __restrict__ ln_g,
    const float* __restrict__ ln_b, float* __restrict__ feat_f32,
    bf16_t* __restrict__ feat_bf) {
  int lane = threadIdx.x & 31;
  int wave = threadIdx.x >> 5;
  int mod = blockIdx.y;
  int col0 = (blockIdx.x * 4 + wave) * 16;
  const float* src = mod ? evt : rgb;
  float* outf = feat_f32 + (size_t)mod * CH * NPIX;
  bf16_t* outb = feat_bf + (size_t)mod * CH * NPIX;

  int n = col0 + (lane & 15);
  int bb = n >> 14;
  int p = n & (HW - 1);
  int mrow = ((lane >> 4) << 3);

  v8f acc[8];
#pragma unroll
  for (int t = 0; t < 8; ++t) acc[t] = zero8();

#pragma unroll
  for (int kt = 0; kt < 2; ++kt) {
    int ib = kt * 32 + ((lane >> 4) << 4);  // input channel base for this lane
    v16bf bfrag;
#pragma unroll
    for (int j = 0; j < 16; ++j)
      bfrag[j] = (bf16_t)src[(size_t)(bb * CIN + ib + j) * HW + p];
#pragma unroll
    for (int mt = 0; mt < 8; ++mt) {
      v16bf a = load_a(w_in_bf, CIN, mt, kt * 32, lane);
      acc[mt] = wmma_bf16(a, bfrag, acc[mt]);
    }
  }

  // LayerNorm over the 128 channels of each pixel column
  float s = 0.f, s2 = 0.f;
#pragma unroll
  for (int t = 0; t < 8; ++t)
#pragma unroll
    for (int r = 0; r < 8; ++r) {
      float v = acc[t][r];
      s += v;
      s2 += v * v;
    }
  s += __shfl_xor(s, 16);
  s2 += __shfl_xor(s2, 16);
  float mean = s * (1.f / 128.f);
  float var = s2 * (1.f / 128.f) - mean * mean;
  float rstd = rsqrtf(var + 1e-5f);

  float* po = outf + (size_t)n * CH + mrow;
  bf16_t* pb = outb + (size_t)n * CH + mrow;
#pragma unroll
  for (int t = 0; t < 8; ++t)
#pragma unroll
    for (int r = 0; r < 8; ++r) {
      int m = t * 16 + mrow + r;
      float y = (acc[t][r] - mean) * rstd * ln_g[m] + ln_b[m];
      po[t * 16 + r] = y;        // 8 consecutive f32 -> b128 stores
      pb[t * 16 + r] = (bf16_t)y;
    }
}

// ---------------------------------------------------------------------------
// Kernel B: hr = relu(conv3x3(w_df1, feat))  as 9-tap implicit GEMM
// ---------------------------------------------------------------------------
__global__ __launch_bounds__(128) void k_conv3_relu(
    const bf16_t* __restrict__ feat_bf, const bf16_t* __restrict__ wdf1_bf,
    bf16_t* __restrict__ hr_bf) {
  int lane = threadIdx.x & 31;
  int wave = threadIdx.x >> 5;
  int mod = blockIdx.y;
  int col0 = (blockIdx.x * 4 + wave) * 16;
  const bf16_t* fb = feat_bf + (size_t)mod * CH * NPIX;
  bf16_t* hb = hr_bf + (size_t)mod * CH * NPIX;

  int n = col0 + (lane & 15);
  int h = (n >> 7) & 127;
  int w = n & 127;
  int mrow = ((lane >> 4) << 3);

  v8f acc[8];
#pragma unroll
  for (int t = 0; t < 8; ++t) acc[t] = zero8();

  for (int tap = 0; tap < 9; ++tap) {
    int dh = tap / 3 - 1, dw = tap % 3 - 1;
    bool valid = ((unsigned)(h + dh) < 128u) && ((unsigned)(w + dw) < 128u);
    int ns = n + dh * WDIM + dw;
    const bf16_t* wt = wdf1_bf + (size_t)tap * (CH * CH);
#pragma unroll
    for (int kt = 0; kt < 4; ++kt) {
      v16bf bfrag;
      if (valid) {
        bfrag = load_b_px(fb, ns, kt, lane);
      } else {
#pragma unroll
        for (int j = 0; j < 16; ++j) bfrag[j] = (bf16_t)0.f;
      }
#pragma unroll
      for (int mt = 0; mt < 8; ++mt) {
        v16bf a = load_a(wt, CH, mt, kt * 32, lane);
        acc[mt] = wmma_bf16(a, bfrag, acc[mt]);
      }
    }
  }

  bf16_t* ph = hb + (size_t)n * CH + mrow;
#pragma unroll
  for (int t = 0; t < 8; ++t)
#pragma unroll
    for (int r = 0; r < 8; ++r) {
      float v = acc[t][r];
      v = v > 0.f ? v : 0.f;
      ph[t * 16 + r] = (bf16_t)v;  // contiguous -> b128 stores
    }
}

// ---------------------------------------------------------------------------
// Kernel C: fused dynamic-filter generation + application.
//   enh[n,c] = sum_{k=0..24} (W_k @ hr)[c,n] * feat[shift_k(n), c]
// hr B-fragments live in registers (reused for all 25 taps). W_k is staged
// into LDS by the TDM (double-buffered, overlapped with compute, row-padded
// against bank conflicts) when the toolchain exposes the builtin; otherwise a
// cooperative padded copy. The 420MB/modality filter tensor never exists.
// ---------------------------------------------------------------------------
__global__ __launch_bounds__(128) void k_dynfilt(
    const bf16_t* __restrict__ hr_bf, const bf16_t* __restrict__ wdf2_bf,
    const float* __restrict__ feat_f32, float* __restrict__ enh_f32,
    bf16_t* __restrict__ enh_bf) {
#if HAVE_TDM
  __shared__ __align__(16) bf16_t lwk[2][CH * LDSLD];  // 2 x 34KB ping-pong
#else
  __shared__ __align__(16) bf16_t lwk[1][CH * LDSLD];
#endif
  int lane = threadIdx.x & 31;
  int wave = threadIdx.x >> 5;
  int mod = blockIdx.y;
  int col0 = (blockIdx.x * 4 + wave) * 16;
  const bf16_t* hb = hr_bf + (size_t)mod * CH * NPIX;
  const float* ff = feat_f32 + (size_t)mod * CH * NPIX;
  float* ef = enh_f32 + (size_t)mod * CH * NPIX;
  bf16_t* eb = enh_bf + (size_t)mod * CH * NPIX;

  int n = col0 + (lane & 15);
  int h = (n >> 7) & 127;
  int w = n & 127;
  int mrow = ((lane >> 4) << 3);

  // Preload the hr B-fragments once; reused by all 25 tap GEMMs.
  v16bf bh[4];
#pragma unroll
  for (int kt = 0; kt < 4; ++kt) bh[kt] = load_b_px(hb, n, kt, lane);

  v8f outa[8];
#pragma unroll
  for (int t = 0; t < 8; ++t) outa[t] = zero8();

#if HAVE_TDM
  if (wave == 0) tdm_load_wk(wdf2_bf, lwk[0]);  // prefetch W_0
#endif

  for (int k = 0; k < 25; ++k) {
#if HAVE_TDM
    if (wave == 0) __builtin_amdgcn_s_wait_tensorcnt(0);  // W_k landed
    __syncthreads();  // everyone sees W_k; everyone done reading W_{k-1}'s buf
    if (wave == 0 && k + 1 < 25)
      tdm_load_wk(wdf2_bf + (size_t)(k + 1) * (CH * CH), lwk[(k + 1) & 1]);
    const bf16_t* wk = lwk[k & 1];
#else
    __syncthreads();
    {  // cooperative padded copy of W_k: 2048 uint4, 16 per row of 17 slots
      const uint4* srcw = (const uint4*)(wdf2_bf + (size_t)k * (CH * CH));
      uint4* dstw = (uint4*)lwk[0];
#pragma unroll
      for (int t = 0; t < 16; ++t) {
        int v = threadIdx.x + t * 128;
        dstw[(v >> 4) * 17 + (v & 15)] = srcw[v];
      }
    }
    __syncthreads();
    const bf16_t* wk = lwk[0];
#endif

    int dh = k / 5 - 2, dw = k % 5 - 2;
    bool valid = ((unsigned)(h + dh) < 128u) && ((unsigned)(w + dw) < 128u);
    int ns = n + dh * WDIM + dw;

#pragma unroll
    for (int mt = 0; mt < 8; ++mt) {
      v8f g = zero8();
#pragma unroll
      for (int kt = 0; kt < 4; ++kt) {
        v16bf a = load_a(wk, LDSLD, mt, kt * 32, lane);
        g = wmma_bf16(a, bh[kt], g);
      }
      if (valid) {
        const float* pf = ff + (size_t)ns * CH + mt * 16 + mrow;
#pragma unroll
        for (int r = 0; r < 8; ++r) outa[mt][r] += g[r] * pf[r];
      }
    }
  }

  float* pe = ef + (size_t)n * CH + mrow;
  bf16_t* pb = eb + (size_t)n * CH + mrow;
#pragma unroll
  for (int t = 0; t < 8; ++t)
#pragma unroll
    for (int r = 0; r < 8; ++r) {
      float v = outa[t][r];
      pe[t * 16 + r] = v;
      pb[t * 16 + r] = (bf16_t)v;
    }
}

// ---------------------------------------------------------------------------
// Kernel D: gate + residual + LN + output projection.
//   s = sigmoid(w_sp @ event_enh); fused = rgb_enh*(1+s) + rgb_feat;
//   out = w_out @ LN(fused)
// LDS relayouts the normalized tile (pixel-major, padded) between the GEMMs.
// ---------------------------------------------------------------------------
#define LTLD 136  // padded per-pixel LDS stride
__global__ __launch_bounds__(128) void k_fuse_out(
    const float* __restrict__ enh_f32, const bf16_t* __restrict__ enh_bf,
    const float* __restrict__ feat_f32, const bf16_t* __restrict__ wsp_bf,
    const bf16_t* __restrict__ wout_bf, const float* __restrict__ ln_g,
    const float* __restrict__ ln_b, float* __restrict__ out) {
  __shared__ __align__(16) bf16_t lt[4][16 * LTLD];  // per-wave norm tile
  int lane = threadIdx.x & 31;
  int wave = threadIdx.x >> 5;
  int col0 = (blockIdx.x * 4 + wave) * 16;

  const float* rgb_enh = enh_f32;                         // mod 0
  const bf16_t* evt_enh_bf = enh_bf + (size_t)CH * NPIX;  // mod 1
  const float* rgb_feat = feat_f32;                       // mod 0

  int n = col0 + (lane & 15);
  int mrow = ((lane >> 4) << 3);

  // s = w_sp @ event_enh
  v8f sacc[8];
#pragma unroll
  for (int t = 0; t < 8; ++t) sacc[t] = zero8();
#pragma unroll
  for (int kt = 0; kt < 4; ++kt) {
    v16bf bfrag = load_b_px(evt_enh_bf, n, kt, lane);
#pragma unroll
    for (int mt = 0; mt < 8; ++mt) {
      v16bf a = load_a(wsp_bf, CH, mt, kt * 32, lane);
      sacc[mt] = wmma_bf16(a, bfrag, sacc[mt]);
    }
  }

  float fus[8][8];
  float s = 0.f, s2 = 0.f;
  const float* pe = rgb_enh + (size_t)n * CH + mrow;
  const float* pf = rgb_feat + (size_t)n * CH + mrow;
#pragma unroll
  for (int t = 0; t < 8; ++t)
#pragma unroll
    for (int r = 0; r < 8; ++r) {
      float sig = 1.f / (1.f + __expf(-sacc[t][r]));
      float re = pe[t * 16 + r];
      float v = re + re * sig + pf[t * 16 + r];
      fus[t][r] = v;
      s += v;
      s2 += v * v;
    }
  s += __shfl_xor(s, 16);
  s2 += __shfl_xor(s2, 16);
  float mean = s * (1.f / 128.f);
  float var = s2 * (1.f / 128.f) - mean * mean;
  float rstd = rsqrtf(var + 1e-5f);

  // normalized tile -> LDS, pixel-major [col][c] (padded stride)
  bf16_t* lw = &lt[wave][(lane & 15) * LTLD + mrow];
#pragma unroll
  for (int t = 0; t < 8; ++t)
#pragma unroll
    for (int r = 0; r < 8; ++r) {
      int m = t * 16 + mrow + r;
      float y = (fus[t][r] - mean) * rstd * ln_g[m] + ln_b[m];
      lw[t * 16 + r] = (bf16_t)y;  // 8 contiguous bf16 -> ds_store_b128
    }
  __syncthreads();

  // out = w_out[64x128] @ norm
  v8f oacc[4];
#pragma unroll
  for (int t = 0; t < 4; ++t) oacc[t] = zero8();
#pragma unroll
  for (int kt = 0; kt < 4; ++kt) {
    const bf16_t* pp =
        &lt[wave][(lane & 15) * LTLD + kt * 32 + ((lane >> 4) << 4)];
    v16bf bfrag;
#pragma unroll
    for (int j = 0; j < 16; ++j) bfrag[j] = pp[j];  // 2x ds_load_b128
#pragma unroll
    for (int mt = 0; mt < 4; ++mt) {
      v16bf a = load_a(wout_bf, CH, mt, kt * 32, lane);
      oacc[mt] = wmma_bf16(a, bfrag, oacc[mt]);
    }
  }

  int bb = n >> 14;
  int p = n & (HW - 1);
#pragma unroll
  for (int t = 0; t < 4; ++t)
#pragma unroll
    for (int r = 0; r < 8; ++r) {
      int c = t * 16 + mrow + r;
      out[(size_t)(bb * CIN + c) * HW + p] = oacc[t][r];
    }
}

// ---------------------------------------------------------------------------
extern "C" void kernel_launch(void* const* d_in, const int* in_sizes, int n_in,
                              void* d_out, int out_size, void* d_ws,
                              size_t ws_size, hipStream_t stream) {
  const float* rgb = (const float*)d_in[0];
  const float* evt = (const float*)d_in[1];
  const float* w_in = (const float*)d_in[2];
  const float* ln_g = (const float*)d_in[3];
  const float* ln_b = (const float*)d_in[4];
  const float* w_df1 = (const float*)d_in[5];
  const float* w_df2 = (const float*)d_in[6];
  const float* w_sp = (const float*)d_in[7];
  const float* w_out = (const float*)d_in[8];

  char* ws = (char*)d_ws;
  size_t off = 0;
  auto alloc = [&](size_t bytes) -> void* {
    void* p = ws + off;
    off = (off + bytes + 255) & ~(size_t)255;
    return p;
  };

  bf16_t* wi = (bf16_t*)alloc((size_t)CH * CIN * 2);
  bf16_t* w1 = (bf16_t*)alloc((size_t)9 * CH * CH * 2);
  bf16_t* w2 = (bf16_t*)alloc((size_t)25 * CH * CH * 2);
  bf16_t* wsb = (bf16_t*)alloc((size_t)CH * CH * 2);
  bf16_t* wo = (bf16_t*)alloc((size_t)CIN * CH * 2);
  float* feat_f32 = (float*)alloc((size_t)2 * CH * NPIX * 4);
  bf16_t* feat_bf = (bf16_t*)alloc((size_t)2 * CH * NPIX * 2);
  bf16_t* hr_bf = (bf16_t*)alloc((size_t)2 * CH * NPIX * 2);
  float* enh_f32 = (float*)alloc((size_t)2 * CH * NPIX * 4);
  bf16_t* enh_bf = (bf16_t*)alloc((size_t)2 * CH * NPIX * 2);
  (void)ws_size;
  (void)in_sizes;
  (void)n_in;
  (void)out_size;

  const int n_w = CH * CIN + 9 * CH * CH + 25 * CH * CH + CH * CH + CIN * CH;
  k_prep_weights<<<(n_w + 255) / 256, 256, 0, stream>>>(
      w_in, w_df1, w_df2, w_sp, w_out, wi, w1, w2, wsb, wo);

  dim3 g2(NPIX / 16 / 4, 2);  // (512, 2 modalities), 4 waves/block
  k_in_ln<<<g2, 128, 0, stream>>>(rgb, evt, wi, ln_g, ln_b, feat_f32, feat_bf);
  k_conv3_relu<<<g2, 128, 0, stream>>>(feat_bf, w1, hr_bf);
  k_dynfilt<<<g2, 128, 0, stream>>>(hr_bf, w2, feat_f32, enh_f32, enh_bf);
  k_fuse_out<<<NPIX / 16 / 4, 128, 0, stream>>>(
      enh_f32, enh_bf, feat_f32, wsb, wo, ln_g, ln_b, (float*)d_out);
}
